// SequenceLoss_17188459118667
// MI455X (gfx1250) — matrix operand order repaired and checked by reading, not compiled, not run on hardware
//
#include <hip/hip_runtime.h>
#include <hip/hip_bf16.h>

typedef __attribute__((ext_vector_type(16))) _Float16 v16h;
typedef __attribute__((ext_vector_type(8)))  float    v8f;

#define ROWH   24              // halfs per LDS row slot (48 B: 16B-aligned, bank-spread)
#define SLOT4  3               // uint4 per row slot (48 B)
#define NSLOTS 33              // rows r .. r+32
#define WPB    8               // waves per block
#define NBLK   512
#define NTHR   256
#define TOTAL_WAVES (NBLK * WPB)

// Per-lane softmax of one row of 16 floats -> 16 halfs packed in two uint4.
// Rows >= T produce zeros (so out-of-range pairs contribute nothing).
__device__ __forceinline__ void softmax16_f16(const float* __restrict__ in,
                                              long long row, long long T,
                                              uint4* lo, uint4* hi) {
  long long rr = row < T ? row : (T - 1);
  const float4* p = (const float4*)(in + rr * 16);
  float4 q0 = p[0], q1 = p[1], q2 = p[2], q3 = p[3];
  float x[16] = {q0.x, q0.y, q0.z, q0.w, q1.x, q1.y, q1.z, q1.w,
                 q2.x, q2.y, q2.z, q2.w, q3.x, q3.y, q3.z, q3.w};
  float mx = x[0];
#pragma unroll
  for (int i = 1; i < 16; ++i) mx = fmaxf(mx, x[i]);
  float e[16];
  float s = 0.f;
#pragma unroll
  for (int i = 0; i < 16; ++i) { e[i] = __expf(x[i] - mx); s += e[i]; }
  float sc = (row < T) ? (1.0f / s) : 0.0f;
  union U { _Float16 h[8]; uint4 u; };
  U a, b;
#pragma unroll
  for (int i = 0; i < 8; ++i) a.h[i] = (_Float16)(e[i] * sc);
#pragma unroll
  for (int i = 0; i < 8; ++i) b.h[i] = (_Float16)(e[i + 8] * sc);
  *lo = a.u;
  *hi = b.u;
}

__global__ void __launch_bounds__(NTHR)
seqloss_wmma_kernel(const float* __restrict__ input, const float* __restrict__ A_mat,
                    float* __restrict__ ws, long long T, int n_chunks, int cpw) {
  __shared__ uint4 lds[WPB * NSLOTS * SLOT4];

  const int lane  = threadIdx.x & 31;
  const int wv    = threadIdx.x >> 5;
  const int gwave = blockIdx.x * WPB + wv;

  uint4*    base4 = &lds[wv * NSLOTS * SLOT4];
  _Float16* baseh = (_Float16*)base4;

  const int m = lane & 15;   // class index for A-row / B-col fragments
  const int h = lane >> 4;   // half-wave selector

  v8f acc = {};              // persistent 16x16 f32 accumulator (C/D fragment)

  int cBeg = gwave * cpw;
  int cEnd = cBeg + cpw;
  if (cEnd > n_chunks) cEnd = n_chunks;

  if (cBeg < cEnd) {
    // Prime slot 0 with row cBeg*32 (all lanes compute it; lane 0 stores).
    uint4 lo, hi;
    softmax16_f16(input, (long long)cBeg * 32, T, &lo, &hi);
    if (lane == 0) { base4[0] = lo; base4[1] = hi; }

    for (int c = cBeg; c < cEnd; ++c) {
      const long long r = (long long)c * 32;

      // Prefetch this lane's row of the NEXT chunk (streaming hint -> global_prefetch_b8).
      {
        long long pr = r + 33 + lane;
        if (pr >= T) pr = T - 1;
        __builtin_prefetch(input + pr * 16, 0, 3);
      }

      // Lane L -> softmax of row r+1+L into slot 1+L (slots 1..32).
      softmax16_f16(input, r + 1 + lane, T, &lo, &hi);
      base4[(1 + lane) * SLOT4 + 0] = lo;
      base4[(1 + lane) * SLOT4 + 1] = hi;
      asm volatile("s_wait_dscnt 0" ::: "memory");  // stores visible before gathers

      // A fragment: A[m][k] = yhat[r+k][m]  (16x32, ISA 16-bit A layout)
      v16h a, b;
#pragma unroll
      for (int v = 0; v < 8; ++v) {
        const int kb = (v < 4 ? 2 * v : 8 + 2 * v) + 8 * h;
        a[2 * v]     = baseh[(kb)     * ROWH + m];
        a[2 * v + 1] = baseh[(kb + 1) * ROWH + m];
      }
      // B fragment: B[k][n] = yhat[r+1+k][n]  (32x16; half e holds K = (e < 8 ? e : 8 + e) + 8h)
#pragma unroll
      for (int e = 0; e < 16; ++e) {
        b[e] = baseh[((e < 8 ? e : 8 + e) + 8 * h + 1) * ROWH + m];
      }

      acc = __builtin_amdgcn_wmma_f32_16x16x32_f16(
          /*neg_a=*/false, a, /*neg_b=*/false, b,
          /*c_mod=*/(short)0, acc, /*reuse_a=*/false, /*reuse_b=*/false);

      // Roll row r+32 (lane 31's row, still in regs) into slot 0 for next chunk.
      asm volatile("" ::: "memory");
      if (lane == 31) { base4[0] = lo; base4[1] = hi; }
    }
  }

  // Weight by log(A) and reduce.  C/D layout: VGPR i, lanes 0-15 -> (M=i, N=lane),
  // lanes 16-31 -> (M=8+i, N=lane-16).
  float s = 0.0f;
  const int mb = h * 8;
#pragma unroll
  for (int i = 0; i < 8; ++i) {
    s += acc[i] * __logf(A_mat[(mb + i) * 16 + m]);
  }
#pragma unroll
  for (int off = 16; off > 0; off >>= 1) s += __shfl_xor(s, off, 32);
  if (lane == 0) ws[gwave] = s;
}

__global__ void __launch_bounds__(256)
seqloss_reduce_kernel(const float* __restrict__ ws, float* __restrict__ out, float invT) {
  __shared__ float sm[256];
  float s = 0.0f;
  for (int i = threadIdx.x; i < TOTAL_WAVES; i += 256) s += ws[i];
  sm[threadIdx.x] = s;
  __syncthreads();
  for (int step = 128; step > 0; step >>= 1) {
    if ((int)threadIdx.x < step) sm[threadIdx.x] += sm[threadIdx.x + step];
    __syncthreads();
  }
  if (threadIdx.x == 0) *out = -sm[0] * invT;
}

extern "C" void kernel_launch(void* const* d_in, const int* in_sizes, int n_in,
                              void* d_out, int out_size, void* d_ws, size_t ws_size,
                              hipStream_t stream) {
  const float* input = (const float*)d_in[0];
  // d_in[1] (target) is unused by the reference forward.
  const float* A = (const float*)d_in[2];
  float* out = (float*)d_out;
  float* ws  = (float*)d_ws;   // needs TOTAL_WAVES * 4 = 16 KB

  const long long Tn      = (long long)(in_sizes[0] / 16);
  const long long n_pairs = Tn - 1;
  const int n_chunks = (int)((n_pairs + 31) / 32);
  const int cpw      = (n_chunks + TOTAL_WAVES - 1) / TOTAL_WAVES;
  const float invT   = 1.0f / (float)Tn;

  seqloss_wmma_kernel<<<dim3(NBLK), dim3(NTHR), 0, stream>>>(
      input, A, ws, Tn, n_chunks, cpw);
  seqloss_reduce_kernel<<<dim3(1), dim3(256), 0, stream>>>(ws, out, invT);
}
